// RBFANN_69698729279913
// MI455X (gfx1250) — matrix operand order repaired and checked by reading, not compiled, not run on hardware
//
#include <hip/hip_runtime.h>
#include <hip/hip_bf16.h>

// MI455X / gfx1250, wave32. bf16 WMMA path (v_wmma_f32_16x16x32_bf16).
// Fused per-graph kernel stages gathered node data K-major in LDS so all
// WMMA fragment reads are vector ds_load_b128s.

typedef __attribute__((ext_vector_type(16))) __bf16 v16bf;
typedef __attribute__((ext_vector_type(8)))  float  v8f;

#define NN  50000
#define CCN 64
#define HSN 64
#define DDN 128
#define BBN 512
#define KPAD 40                         // LDS row stride (elems): 80 B, 16B-aligned
#define W1P_ELEMS (4 * 4 * 32 * 16)     // k0-tiles x col-tiles x lanes x elems

// 16-bit A-matrix 16x32 VGPR layout (ISA 7.12.2): element e of lane l -> K index.
__device__ __forceinline__ int kmapA(int e, int half) {
    return (e & 7) + 8 * (2 * (e >> 3) + half);
}
// Contiguous K order (unused; not the hardware table order): lanes 0-15 hold K=0..15 (e=K), lanes 16-31 hold K=16..31.
__device__ __forceinline__ int kmapB(int e, int half) {
    return e + 16 * half;
}

union V16 { uint4 q[2]; v16bf v; };     // assemble fragment from 2 LDS b128 reads
union Pack4 { __bf16 h[4]; uint2 u; };  // 4 bf16 -> one ds_store_b64

// ---------------------------------------------------------------------------
// K0: zero the per-graph histogram
// ---------------------------------------------------------------------------
__global__ void zero_counts_kernel(int* __restrict__ counts) {
    counts[threadIdx.x] = 0;
}

// ---------------------------------------------------------------------------
// K0b: pre-pack W1 (f32 [128,64]) into bf16 WMMA B-fragment layout so each
// lane's 16 B elements are one contiguous 32-byte vector (L2-resident, 16 KB).
// ---------------------------------------------------------------------------
__global__ void pack_W1_kernel(const float* __restrict__ W1,
                               __bf16* __restrict__ W1p) {
    const int t = blockIdx.x * blockDim.x + threadIdx.x;   // 0..8191
    if (t >= W1P_ELEMS) return;
    const int e    = t & 15;
    const int lane = (t >> 4) & 31;
    const int cbt  = (t >> 9) & 3;
    const int k0t  = t >> 11;
    const int half = lane >> 4;
    const int col  = lane & 15;
    W1p[t] = (__bf16)W1[(size_t)(k0t * 32 + kmapA(e, half)) * HSN + cbt * 16 + col];
}

// ---------------------------------------------------------------------------
// K1: x1 = bf16( x @ W1 + b1 )   [N,64], via WMMA bf16.
// grid = N/16 row tiles, block = 128 threads (4 waves = 4 column tiles of 16)
// ---------------------------------------------------------------------------
__global__ void x1_wmma_kernel(const float* __restrict__ x,
                               const __bf16* __restrict__ W1p,
                               const float* __restrict__ b1,
                               __bf16* __restrict__ x1b) {
    const int wave = threadIdx.x >> 5;
    const int lane = threadIdx.x & 31;
    const int half = lane >> 4;        // 0: lanes 0-15, 1: lanes 16-31
    const int col  = lane & 15;
    const int r16  = blockIdx.x * 16;  // row tile base
    const int cb   = wave * 16;        // col tile base (0,16,32,48)

    v8f acc = {};
    const float* xrow = x + (size_t)(r16 + col) * DDN;   // A row owned by lane
    const v16bf* W1pv = (const v16bf*)W1p;

    #pragma unroll
    for (int k0 = 0; k0 < DDN; k0 += 32) {
        v16bf a;
        // A: two runs of 8 consecutive K values -> vectorized float4 loads
        #pragma unroll
        for (int q = 0; q < 2; q++) {
            const int kb = k0 + 8 * (2 * q + half);
            const float4 u = *(const float4*)(xrow + kb);
            const float4 v = *(const float4*)(xrow + kb + 4);
            a[8 * q + 0] = (__bf16)u.x; a[8 * q + 1] = (__bf16)u.y;
            a[8 * q + 2] = (__bf16)u.z; a[8 * q + 3] = (__bf16)u.w;
            a[8 * q + 4] = (__bf16)v.x; a[8 * q + 5] = (__bf16)v.y;
            a[8 * q + 6] = (__bf16)v.z; a[8 * q + 7] = (__bf16)v.w;
        }
        // B: one 32-byte vector load from the pre-packed fragment buffer
        const v16bf bm = W1pv[((k0 >> 5) * 4 + wave) * 32 + lane];

        acc = __builtin_amdgcn_wmma_f32_16x16x32_bf16(
                  false, a, false, bm, (short)0, acc, false, false);
    }

    const float bias = b1[cb + col];
    #pragma unroll
    for (int r = 0; r < 8; r++) {
        const int mm = r + 8 * half;  // C/D layout: M = vgpr + 8*half, N = col
        x1b[(size_t)(r16 + mm) * HSN + cb + col] = (__bf16)(acc[r] + bias);
    }
}

// ---------------------------------------------------------------------------
// K2: histogram of batch ids
// ---------------------------------------------------------------------------
__global__ void hist_kernel(const long long* __restrict__ batch,
                            int* __restrict__ counts, int n) {
    int i = blockIdx.x * blockDim.x + threadIdx.x;
    if (i < n) atomicAdd(&counts[(int)batch[i]], 1);
}

// ---------------------------------------------------------------------------
// K3: exclusive scan over 512 counts (single block Hillis-Steele)
// ---------------------------------------------------------------------------
__global__ void scan_kernel(const int* __restrict__ counts,
                            int* __restrict__ offsets,
                            int* __restrict__ cursor) {
    __shared__ int tmp[BBN];
    const int t = threadIdx.x;
    tmp[t] = counts[t];
    __syncthreads();
    for (int off = 1; off < BBN; off <<= 1) {
        int v = (t >= off) ? tmp[t - off] : 0;
        __syncthreads();
        tmp[t] += v;
        __syncthreads();
    }
    const int excl = tmp[t] - counts[t];
    offsets[t] = excl;
    cursor[t]  = excl;
}

// ---------------------------------------------------------------------------
// K4: scatter node indices grouped by graph (counting sort)
// ---------------------------------------------------------------------------
__global__ void scatter_kernel(const long long* __restrict__ batch,
                               int* __restrict__ cursor,
                               int* __restrict__ sorted, int n) {
    int i = blockIdx.x * blockDim.x + threadIdx.x;
    if (i < n) {
        int b = (int)batch[i];
        int p = atomicAdd(&cursor[b], 1);
        sorted[p] = i;
    }
}

// ---------------------------------------------------------------------------
// K5: fused per-graph kernel. One block (256 thr = 8 waves) per graph:
//   feat[b][64][64] = sum_i rbf_i^T (x) x1_i  via WMMA bf16, K-chunks of 32.
//   Chunk data staged K-major (transposed) in LDS: each wave owns 4 nodes,
//   each lane an element pair -> coalesced global reads, packed ds_store_b64
//   scatter, and all WMMA fragments read as aligned ds_load_b128s.
//   Then attention softmax over centers, MLP head, LayerNorm, final dot.
//   feat never leaves the WGP (accumulators -> LDS).
// ---------------------------------------------------------------------------
__global__ void fused_graph_kernel(const float* __restrict__ pos,
                                   const float* __restrict__ centers,
                                   const float* __restrict__ widths,
                                   const __bf16* __restrict__ x1b,
                                   const int* __restrict__ sorted,
                                   const int* __restrict__ offsets,
                                   const int* __restrict__ counts,
                                   const float* __restrict__ Wa,
                                   const float* __restrict__ ba,
                                   const float* __restrict__ W2,
                                   const float* __restrict__ b2,
                                   const float* __restrict__ gamma,
                                   const float* __restrict__ beta,
                                   const float* __restrict__ W3,
                                   const float* __restrict__ b3,
                                   float* __restrict__ out) {
    __shared__ __align__(16) __bf16 rbf_t[CCN][KPAD];  // [center][node]  5 KB
    __shared__ __align__(16) __bf16 x1_t[HSN][KPAD];   // [hidden][node]  5 KB
    __shared__ float feat_s[CCN][HSN];                 // 16 KB
    __shared__ float wrow[CCN];
    __shared__ float attn[CCN];
    __shared__ float aggs[HSN];
    __shared__ float hvec[DDN];
    __shared__ float red[256];
    __shared__ float stats[2];

    const int b     = blockIdx.x;
    const int start = offsets[b];
    const int cnt   = counts[b];

    const int wave = threadIdx.x >> 5;   // == node-quad owner (sq)
    const int lane = threadIdx.x & 31;   // == element-pair owner (eg)
    const int half = lane >> 4;
    const int col  = lane & 15;

    // 16 output tiles (4x4 of 16x16) over 8 waves; tiles 2w and 2w+1 share A.
    const int tc  = wave >> 1;
    const int th0 = (2 * wave) & 3;
    const int th1 = (2 * wave + 1) & 3;

    v8f acc0 = {}, acc1 = {};

    for (int chunk = 0; chunk < cnt; chunk += 32) {
        // prefetch next chunk's gather indices (global_prefetch_b8)
        if (chunk + 32 + lane < cnt)
            __builtin_prefetch(&sorted[start + chunk + 32 + lane], 0, 1);

        // ---- stage: wave owns nodes 4*wave..4*wave+3, lane owns elems 2l,2l+1
        Pack4 pr[2], px4[2];             // [t]: center/hidden = 2*lane + t
        #pragma unroll
        for (int i = 0; i < 4; i++) {
            const int j = chunk + 4 * wave + i;
            if (j < cnt) {
                const int idx = sorted[start + j];
                // x1 element pair: one coalesced b32 (full row per wave)
                union { unsigned u; __bf16 h[2]; } xw;
                xw.u = *(const unsigned*)(x1b + (size_t)idx * HSN + 2 * lane);
                px4[0].h[i] = xw.h[0];
                px4[1].h[i] = xw.h[1];
                // rbf on the fly (each (node,center) computed exactly once)
                const float px = pos[idx * 3 + 0];
                const float py = pos[idx * 3 + 1];
                const float pz = pos[idx * 3 + 2];
                #pragma unroll
                for (int t = 0; t < 2; t++) {
                    const int c = 2 * lane + t;
                    const float dx = px - centers[c * 3 + 0];
                    const float dy = py - centers[c * 3 + 1];
                    const float dz = pz - centers[c * 3 + 2];
                    const float dist = sqrtf(dx * dx + dy * dy + dz * dz);
                    const float w = widths[c];
                    pr[t].h[i] = (__bf16)__expf(-dist / (w * w));
                }
            } else {
                px4[0].h[i] = (__bf16)0.0f; px4[1].h[i] = (__bf16)0.0f;
                pr[0].h[i]  = (__bf16)0.0f; pr[1].h[i]  = (__bf16)0.0f;
            }
        }
        #pragma unroll
        for (int t = 0; t < 2; t++) {    // packed transposed scatter: ds_store_b64
            *(uint2*)&rbf_t[2 * lane + t][4 * wave] = pr[t].u;
            *(uint2*)&x1_t[2 * lane + t][4 * wave] = px4[t].u;
        }
        __syncthreads();

        // ---- WMMA fragments: contiguous K runs -> aligned ds_load_b128 pairs
        V16 av, bv0, bv1;
        const int ar  = tc * 16 + col;
        av.q[0]  = *(const uint4*)&rbf_t[ar][16 * half];       // k = 16h+0..7
        av.q[1]  = *(const uint4*)&rbf_t[ar][16 * half + 8];   // k = 16h+8..15
        const int br0 = th0 * 16 + col;
        const int br1 = th1 * 16 + col;
        bv0.q[0] = *(const uint4*)&x1_t[br0][16 * half];       // k = 16h+0..7
        bv0.q[1] = *(const uint4*)&x1_t[br0][16 * half + 8];   // k = 16h+8..15
        bv1.q[0] = *(const uint4*)&x1_t[br1][16 * half];
        bv1.q[1] = *(const uint4*)&x1_t[br1][16 * half + 8];

        acc0 = __builtin_amdgcn_wmma_f32_16x16x32_bf16(
                   false, av.v, false, bv0.v, (short)0, acc0, false, false);
        acc1 = __builtin_amdgcn_wmma_f32_16x16x32_bf16(
                   false, av.v, false, bv1.v, (short)0, acc1, false, false);
        __syncthreads();
    }

    // spill feat tiles to LDS (C/D layout: M = vgpr + 8*half, N = col)
    #pragma unroll
    for (int r = 0; r < 8; r++) {
        const int m = r + 8 * half;
        feat_s[tc * 16 + m][th0 * 16 + col] = acc0[r];
        feat_s[tc * 16 + m][th1 * 16 + col] = acc1[r];
    }
    __syncthreads();

    // ---- attention logits over centers ----
    if (threadIdx.x < CCN) {
        const int c = threadIdx.x;
        float sum = ba[0];
        #pragma unroll 8
        for (int h = 0; h < HSN; h++) sum += feat_s[c][h] * Wa[h];
        wrow[c] = sum;
    }
    __syncthreads();
    if (threadIdx.x == 0) {
        float mx = wrow[0];
        for (int c = 1; c < CCN; c++) mx = fmaxf(mx, wrow[c]);
        float sum = 0.0f;
        for (int c = 0; c < CCN; c++) { float e = __expf(wrow[c] - mx); attn[c] = e; sum += e; }
        const float inv = 1.0f / sum;
        for (int c = 0; c < CCN; c++) attn[c] *= inv;
    }
    __syncthreads();
    // ---- agg[h] = sum_c attn[c] * feat[c][h] ----
    if (threadIdx.x < HSN) {
        const int h = threadIdx.x;
        float sum = 0.0f;
        #pragma unroll 8
        for (int c = 0; c < CCN; c++) sum += attn[c] * feat_s[c][h];
        aggs[h] = sum;
    }
    __syncthreads();
    // ---- MLP: h = LeakyReLU(agg @ W2 + b2) ----
    if (threadIdx.x < DDN) {
        const int d = threadIdx.x;
        float sum = b2[d];
        #pragma unroll 8
        for (int h = 0; h < HSN; h++) sum += aggs[h] * W2[(size_t)h * DDN + d];
        hvec[d] = (sum >= 0.0f) ? sum : 0.2f * sum;
    }
    __syncthreads();
    // ---- LayerNorm stats ----
    if (threadIdx.x == 0) {
        float mu = 0.0f;
        for (int d = 0; d < DDN; d++) mu += hvec[d];
        mu *= (1.0f / DDN);
        float var = 0.0f;
        for (int d = 0; d < DDN; d++) { float t = hvec[d] - mu; var += t * t; }
        var *= (1.0f / DDN);
        stats[0] = mu;
        stats[1] = rsqrtf(var + 1e-5f);
    }
    __syncthreads();
    // ---- normalize, scale, final dot with W3 ----
    float p = 0.0f;
    if (threadIdx.x < DDN) {
        const int d = threadIdx.x;
        const float hn = (hvec[d] - stats[0]) * stats[1] * gamma[d] + beta[d];
        p = hn * W3[d];
    }
    red[threadIdx.x] = p;
    __syncthreads();
    for (int st = 128; st > 0; st >>= 1) {
        if (threadIdx.x < st) red[threadIdx.x] += red[threadIdx.x + st];
        __syncthreads();
    }
    if (threadIdx.x == 0) out[b] = red[0] + b3[0];
}

// ---------------------------------------------------------------------------
extern "C" void kernel_launch(void* const* d_in, const int* in_sizes, int n_in,
                              void* d_out, int out_size, void* d_ws, size_t ws_size,
                              hipStream_t stream) {
    const float*     x       = (const float*)d_in[0];
    const float*     pos     = (const float*)d_in[1];
    const long long* batch   = (const long long*)d_in[2];
    const float*     centers = (const float*)d_in[3];
    const float*     widths  = (const float*)d_in[4];
    const float*     W1      = (const float*)d_in[5];
    const float*     b1      = (const float*)d_in[6];
    const float*     Wa      = (const float*)d_in[7];
    const float*     ba      = (const float*)d_in[8];
    const float*     W2      = (const float*)d_in[9];
    const float*     b2      = (const float*)d_in[10];
    const float*     gamma   = (const float*)d_in[11];
    const float*     beta    = (const float*)d_in[12];
    const float*     W3      = (const float*)d_in[13];
    const float*     b3      = (const float*)d_in[14];
    float*           out     = (float*)d_out;

    char* ws = (char*)d_ws;
    size_t off = 0;
    __bf16* x1b    = (__bf16*)(ws + off); off += (size_t)NN * HSN * sizeof(__bf16);
    __bf16* W1p    = (__bf16*)(ws + off); off += (size_t)W1P_ELEMS * sizeof(__bf16);
    int*    counts  = (int*)(ws + off);   off += BBN * sizeof(int);
    int*    offsets = (int*)(ws + off);   off += BBN * sizeof(int);
    int*    cursor  = (int*)(ws + off);   off += BBN * sizeof(int);
    int*    sorted  = (int*)(ws + off);   off += (size_t)NN * sizeof(int);
    (void)ws_size; (void)in_sizes; (void)n_in; (void)out_size;

    zero_counts_kernel<<<1, BBN, 0, stream>>>(counts);
    pack_W1_kernel<<<W1P_ELEMS / 256, 256, 0, stream>>>(W1, W1p);
    x1_wmma_kernel<<<NN / 16, 128, 0, stream>>>(x, W1p, b1, x1b);
    hist_kernel<<<(NN + 255) / 256, 256, 0, stream>>>(batch, counts, NN);
    scan_kernel<<<1, BBN, 0, stream>>>(counts, offsets, cursor);
    scatter_kernel<<<(NN + 255) / 256, 256, 0, stream>>>(batch, cursor, sorted, NN);
    fused_graph_kernel<<<BBN, 256, 0, stream>>>(pos, centers, widths, x1b,
                                                sorted, offsets, counts,
                                                Wa, ba, W2, b2, gamma, beta,
                                                W3, b3, out);
}
